// ExpertODEEnsemble_38517266710821
// MI455X (gfx1250) — matrix operand order repaired and checked by reading, not compiled, not run on hardware
//
#include <hip/hip_runtime.h>

#define EXP 8
#define DD 64
#define HH 512
#define BB 32768
#define IN_DIM 67
#define ROWS 128   // batch rows per block

typedef __bf16 bf16t;
typedef __attribute__((ext_vector_type(16))) bf16t          v16bf;
typedef __attribute__((ext_vector_type(8)))  float          v8f;
typedef __attribute__((ext_vector_type(16))) unsigned short v16u;
typedef __attribute__((ext_vector_type(8)))  unsigned short v8u;

// float -> bf16 round-to-nearest-even
__device__ __forceinline__ unsigned short f2bf(float f) {
    unsigned int u = __float_as_uint(f);
    u += 0x7FFFu + ((u >> 16) & 1u);
    return (unsigned short)(u >> 16);
}

// tanh(x) = 1 - 2/(exp(2x)+1); v_mul + v_exp + v_add + v_rcp + v_fma
__device__ __forceinline__ float tanh_fast(float x) {
    float e = __builtin_amdgcn_exp2f(x * 2.8853900817779268f); // 2*log2(e)
    float r = __builtin_amdgcn_rcpf(e + 1.0f);
    return __builtin_fmaf(-2.0f, r, 1.0f);
}

__device__ __forceinline__ v8f wmma_bf16(v16u a, v16u b, v8f c) {
    return __builtin_amdgcn_wmma_f32_16x16x32_bf16(
        false, __builtin_bit_cast(v16bf, a),
        false, __builtin_bit_cast(v16bf, b),
        (short)0, c, false, false);
}

__device__ __forceinline__ v16u ldsAfrag(const unsigned short* p) {
    v8u lo = *(const v8u*)(p);
    v8u hi = *(const v8u*)(p + 16);
    return __builtin_shufflevector(lo, hi, 0, 1, 2, 3, 4, 5, 6, 7,
                                   8, 9, 10, 11, 12, 13, 14, 15);
}

// ---------------- prepass kernels ----------------

__global__ void conv_bf16_kernel(const float* __restrict__ src,
                                 unsigned short* __restrict__ dst, int n) {
    int i = blockIdx.x * blockDim.x + threadIdx.x;
    if (i < n) dst[i] = f2bf(src[i]);
}

// Repack a row-major (N x K) fp32 weight matrix (per expert, source row
// stride srcStride) into fragment-major bf16: one contiguous 1KB block per
// WMMA B-fragment; lane l of fragment (e,nt,kc) holds
//   col n = nt*16 + (l&15), K = kc*32 + (l>>4)*8 + {0..7, 16..23}  (16 bf16)
__global__ void pack_w_kernel(const float* __restrict__ src,
                              unsigned short* __restrict__ dst,
                              int N, int K, int srcStride, int nExp) {
    const int NT = N >> 4, KC = K >> 5;
    const int total = nExp * NT * KC * 32;
    int tid = blockIdx.x * blockDim.x + threadIdx.x;
    if (tid >= total) return;
    const int lane = tid & 31;
    const int kc   = (tid >> 5) % KC;
    const int nt   = ((tid >> 5) / KC) % NT;
    const int e    = tid / (32 * KC * NT);
    const int n  = nt * 16 + (lane & 15);
    const int k0 = kc * 32 + (lane >> 4) * 8;
    const float* s = src + ((size_t)e * N + n) * srcStride + k0;
    unsigned short* d = dst + (size_t)tid * 16;
    #pragma unroll
    for (int i = 0; i < 16; ++i) d[i] = f2bf(s[i < 8 ? i : i + 8]);
}

// fold scalar-t / sin / cos columns of W1 into an effective bias
__global__ void prep_b1_kernel(const float* __restrict__ W1,
                               const float* __restrict__ b1,
                               const float* __restrict__ t,
                               const float* __restrict__ omega,
                               float* __restrict__ b1eff) {
    int idx = blockIdx.x * blockDim.x + threadIdx.x; // e*HH + h
    if (idx >= EXP * HH) return;
    int e = idx / HH;
    const float* row = W1 + (size_t)idx * IN_DIM;
    float tv = t[0];
    float wt = omega[e] * tv;
    b1eff[idx] = b1[idx] + row[64] * tv + row[65] * sinf(wt) + row[66] * cosf(wt);
}

// ---------------- main fused MoE-ODE kernel ----------------
// Block = 8 waves, 128 batch rows. Activations ping-pong between two LDS
// buffers (128x512 bf16). Wave w owns TWO M-strips (rows (w&3)*16 and +64)
// and 16 N-tiles: strip-0 A-frags register-resident per layer, strip-1
// A-frags stream from LDS, every packed B-fragment feeds TWO WMMAs, and the
// B stream is explicitly double-buffered (one-fragment lookahead over the
// flat fragment-major weight stream).

template<int KC>  // K=32 chunks: 2 for layer1, 16 for hidden layers
__device__ __forceinline__ void layer_fwd(const unsigned short* __restrict__ inBuf,
                                          int inStride,
                                          unsigned short* __restrict__ outBuf,
                                          const unsigned short* __restrict__ Wp,
                                          const float* __restrict__ bias,
                                          unsigned wave, unsigned lane,
                                          unsigned laneLo, unsigned laneHi) {
    const unsigned mt = wave & 3u;
    const unsigned ntBase = (wave >> 2) * 16u;

    const unsigned short* arow0 = inBuf + (mt * 16u + laneLo) * inStride + laneHi * 8u;
    const unsigned short* arow1 = arow0 + 64u * inStride;

    // strip-0 A-fragments register-resident for the whole layer
    v16u af[KC];
    #pragma unroll
    for (int kc = 0; kc < KC; ++kc) af[kc] = ldsAfrag(arow0 + kc * 32);

    // flat fragment-major weight stream for this wave (1KB per fragment)
    const unsigned short* wptr = Wp + ((size_t)ntBase * KC) * 512 + lane * 16;
    v16u b = *(const v16u*)(wptr);   // prime the B double-buffer

    for (int nt = 0; nt < 16; ++nt) {
        const int n0 = (int)(ntBase + nt) * 16;
        const float bv = bias[n0 + laneLo];
        v8f c0, c1;
        #pragma unroll
        for (int r = 0; r < 8; ++r) { c0[r] = bv; c1[r] = bv; }
        #pragma unroll
        for (int kc = 0; kc < KC; ++kc) {
            // lookahead: next fragment is always +512 elements in the flat
            // stream (crosses nt boundaries; final read past the wave's
            // region stays inside the workspace and is discarded)
            v16u bn = *(const v16u*)(wptr + 512);
            c0 = wmma_bf16(af[kc], b, c0);
            v16u a1 = ldsAfrag(arow1 + kc * 32);             // strip-1 from LDS
            c1 = wmma_bf16(a1, b, c1);
            b = bn;
            wptr += 512;
        }
        #pragma unroll
        for (int r = 0; r < 8; ++r) {
            const unsigned row0 = mt * 16u + r + laneHi * 8u;
            outBuf[row0 * HH + n0 + laneLo]         = f2bf(tanh_fast(c0[r]));
            outBuf[(row0 + 64u) * HH + n0 + laneLo] = f2bf(tanh_fast(c1[r]));
        }
    }
}

__launch_bounds__(256, 1)
__global__ void moe_ode_kernel(const unsigned short* __restrict__ xbf,
                               const float* __restrict__ ew,
                               const unsigned short* __restrict__ W1p,
                               const float* __restrict__ b1eff,
                               const unsigned short* __restrict__ W2p,
                               const float* __restrict__ b2,
                               const unsigned short* __restrict__ W3p,
                               const float* __restrict__ b3,
                               const unsigned short* __restrict__ W4p,
                               const float* __restrict__ b4,
                               float* __restrict__ out) {
    extern __shared__ char smem[];
    unsigned short* buf0 = (unsigned short*)smem;     // 128*512 bf16 = 128KB
    unsigned short* buf1 = buf0 + ROWS * HH;          // 128KB
    unsigned short* xt   = buf1 + ROWS * HH;          // 128*64 bf16 = 16KB
    float*          ews  = (float*)(xt + ROWS * DD);  // 128*8 f32 = 4KB

    const unsigned wave   = threadIdx.x >> 5;
    const unsigned lane   = threadIdx.x & 31u;
    const unsigned laneLo = lane & 15u;
    const unsigned laneHi = lane >> 4;
    const size_t   blockRow = (size_t)blockIdx.x * ROWS;

    // stage x tile (128x64 bf16) and expert-weight tile (128x8 f32) into LDS
    {
        const unsigned i = threadIdx.x;
        *(v16u*)(xt + i * 32)      = *(const v16u*)(xbf + blockRow * DD + i * 32);
        *(v16u*)(xt + i * 32 + 16) = *(const v16u*)(xbf + blockRow * DD + i * 32 + 16);
        *(float4*)(ews + i * 4)    = *(const float4*)(ew + blockRow * EXP + i * 4);
    }
    __syncthreads();

    // layer-4 accumulators: wave w owns mt ∈ {w&3, 4+(w&3)}, nt ∈ (w>>2)*2+{0,1}
    const unsigned mt4  = wave & 3u;
    const unsigned ntb4 = (wave >> 2) * 2u;
    v8f outacc0[2], outacc1[2];
    #pragma unroll
    for (int j = 0; j < 2; ++j)
        #pragma unroll
        for (int r = 0; r < 8; ++r) { outacc0[j][r] = 0.0f; outacc1[j][r] = 0.0f; }

    for (int e = 0; e < EXP; ++e) {
        layer_fwd<2>(xt, DD, buf0, W1p + (size_t)e * HH * DD, b1eff + (size_t)e * HH,
                     wave, lane, laneLo, laneHi);
        __syncthreads();
        layer_fwd<16>(buf0, HH, buf1, W2p + (size_t)e * HH * HH, b2 + (size_t)e * HH,
                      wave, lane, laneLo, laneHi);
        __syncthreads();
        layer_fwd<16>(buf1, HH, buf0, W3p + (size_t)e * HH * HH, b3 + (size_t)e * HH,
                      wave, lane, laneLo, laneHi);
        __syncthreads();

        // ---- layer 4: (128x512) x (512x64) + b4, weighted combine into regs ----
        {
            const unsigned short* arow0 = buf0 + (mt4 * 16u + laneLo) * HH + laneHi * 8u;
            const unsigned short* arow1 = arow0 + 64u * HH;
            v16u af[16];
            #pragma unroll
            for (int kc = 0; kc < 16; ++kc) af[kc] = ldsAfrag(arow0 + kc * 32);

            const float* b4e = b4 + (size_t)e * DD;
            const unsigned short* wptr = W4p + (size_t)e * DD * HH
                                       + ((size_t)ntb4 * 16) * 512 + lane * 16;
            v16u b = *(const v16u*)(wptr);
            #pragma unroll
            for (int j = 0; j < 2; ++j) {
                const int n0 = (int)(ntb4 + j) * 16;
                const float bv = b4e[n0 + laneLo];
                v8f c0, c1;
                #pragma unroll
                for (int r = 0; r < 8; ++r) { c0[r] = bv; c1[r] = bv; }
                #pragma unroll
                for (int kc = 0; kc < 16; ++kc) {
                    v16u bn = *(const v16u*)(wptr + 512);
                    c0 = wmma_bf16(af[kc], b, c0);
                    v16u a1 = ldsAfrag(arow1 + kc * 32);
                    c1 = wmma_bf16(a1, b, c1);
                    b = bn;
                    wptr += 512;
                }
                #pragma unroll
                for (int r = 0; r < 8; ++r) {
                    const unsigned rowL0 = mt4 * 16u + r + laneHi * 8u;
                    const unsigned rowL1 = rowL0 + 64u;
                    outacc0[j][r] += ews[rowL0 * EXP + e] * c0[r];
                    outacc1[j][r] += ews[rowL1 * EXP + e] * c1[r];
                }
            }
        }
        __syncthreads(); // buf0 free before next expert's layer1 rewrites it
    }

    // ---- store output tiles (C/D layout: row = r + 8*laneHi, col = laneLo) ----
    #pragma unroll
    for (int j = 0; j < 2; ++j)
        #pragma unroll
        for (int r = 0; r < 8; ++r) {
            const unsigned rowL0 = mt4 * 16u + r + laneHi * 8u;
            const unsigned col   = (ntb4 + j) * 16u + laneLo;
            out[(blockRow + rowL0) * DD + col]       = outacc0[j][r];
            out[(blockRow + rowL0 + 64u) * DD + col] = outacc1[j][r];
        }
}

// ---------------- host launcher ----------------

extern "C" void kernel_launch(void* const* d_in, const int* in_sizes, int n_in,
                              void* d_out, int out_size, void* d_ws, size_t ws_size,
                              hipStream_t stream) {
    const float* t   = (const float*)d_in[0];
    const float* x   = (const float*)d_in[1];
    const float* ew  = (const float*)d_in[2];
    const float* om  = (const float*)d_in[3];
    const float* W1  = (const float*)d_in[4];
    const float* b1  = (const float*)d_in[5];
    const float* W2  = (const float*)d_in[6];
    const float* b2  = (const float*)d_in[7];
    const float* W3  = (const float*)d_in[8];
    const float* b3  = (const float*)d_in[9];
    const float* W4  = (const float*)d_in[10];
    const float* b4  = (const float*)d_in[11];
    float* out = (float*)d_out;

    char* ws = (char*)d_ws;
    size_t off = 0;
    auto alloc = [&](size_t bytes) -> char* {
        off = (off + 255) & ~(size_t)255;
        char* p = ws + off;
        off += bytes;
        return p;
    };
    unsigned short* xbf  = (unsigned short*)alloc((size_t)BB * DD * 2);
    unsigned short* W1p  = (unsigned short*)alloc((size_t)EXP * HH * DD * 2);
    unsigned short* W2p  = (unsigned short*)alloc((size_t)EXP * HH * HH * 2);
    unsigned short* W3p  = (unsigned short*)alloc((size_t)EXP * HH * HH * 2);
    unsigned short* W4p  = (unsigned short*)alloc((size_t)EXP * DD * HH * 2);
    float*          b1e  = (float*)alloc((size_t)EXP * HH * 4);
    // extra tail page so the B-stream lookahead's final +1KB read stays in ws
    (void)alloc(2048);
    (void)ws_size; (void)in_sizes; (void)n_in; (void)out_size;

    int n;
    n = BB * DD;
    conv_bf16_kernel<<<(n + 255) / 256, 256, 0, stream>>>(x, xbf, n);

    // weight repack into fragment-major bf16 (one contiguous 1KB per B-frag)
    n = EXP * (HH >> 4) * (DD >> 5) * 32;  // W1: N=512, K=64, srcStride=67
    pack_w_kernel<<<(n + 255) / 256, 256, 0, stream>>>(W1, W1p, HH, DD, IN_DIM, EXP);
    n = EXP * (HH >> 4) * (HH >> 5) * 32;  // W2/W3: N=512, K=512
    pack_w_kernel<<<(n + 255) / 256, 256, 0, stream>>>(W2, W2p, HH, HH, HH, EXP);
    pack_w_kernel<<<(n + 255) / 256, 256, 0, stream>>>(W3, W3p, HH, HH, HH, EXP);
    n = EXP * (DD >> 4) * (HH >> 5) * 32;  // W4: N=64, K=512
    pack_w_kernel<<<(n + 255) / 256, 256, 0, stream>>>(W4, W4p, DD, HH, HH, EXP);

    prep_b1_kernel<<<(EXP * HH + 255) / 256, 256, 0, stream>>>(W1, b1, t, om, b1e);

    // dynamic LDS: 2 x 128KB act buffers + 16KB x tile + 4KB ew tile = 276KB
    const size_t smemBytes = (size_t)(2 * ROWS * HH + ROWS * DD) * sizeof(unsigned short)
                           + (size_t)ROWS * EXP * sizeof(float);
    moe_ode_kernel<<<BB / ROWS, 256, smemBytes, stream>>>(xbf, ew, W1p, b1e,
                                                          W2p, b2, W3p, b3, W4p, b4, out);
}
